// ReformerEncoderLayer_19164144075420
// MI455X (gfx1250) — matrix operand order repaired and checked by reading, not compiled, not run on hardware
//
#include <hip/hip_runtime.h>
#include <cstdint>

// ---------------------------------------------------------------------------
// Reformer encoder layer for MI455X (gfx1250, wave32, WMMA bf16).
// B=8 S=4096 D=512 H=8 DK=DV=64 BUCKET=64 C=64 DFF=2048, ~190 GFLOP/layer.
// All GEMMs (incl. chunked attention matmuls) run on v_wmma_f32_16x16x32_bf16
// with f32 accumulation. GEMM A-tiles are staged with gfx1250 async
// global->LDS copies (ASYNCcnt); B-tiles are transposed through VGPRs.
// Hashing / bitonic sort / softmax / layernorm are VALU+LDS kernels.
// ---------------------------------------------------------------------------

typedef __bf16 bf16x16 __attribute__((ext_vector_type(16)));
typedef float  f32x8   __attribute__((ext_vector_type(8)));
typedef int    i32x8   __attribute__((ext_vector_type(8)));

__device__ __forceinline__ f32x8 zero8() {
  f32x8 z = {0.f, 0.f, 0.f, 0.f, 0.f, 0.f, 0.f, 0.f};
  return z;
}

// round-to-nearest-even f32 -> bf16 bit pattern
__device__ __forceinline__ unsigned short f2bf(float f) {
  unsigned u = __float_as_uint(f);
  unsigned r = u + 0x7FFFu + ((u >> 16) & 1u);
  return (unsigned short)(r >> 16);
}

__device__ __forceinline__ f32x8 wmma_bf16(bf16x16 a, bf16x16 b, f32x8 c) {
  return __builtin_amdgcn_wmma_f32_16x16x32_bf16(false, a, false, b,
                                                 (short)0, c, false, false);
}

// Per-lane async 16B copy global -> LDS (gfx1250, tracked by ASYNCcnt).
__device__ __forceinline__ void async_copy_b128(unsigned lds_off,
                                                const void* gptr) {
  asm volatile("global_load_async_to_lds_b128 %0, %1, off"
               :
               : "v"(lds_off), "v"((unsigned long long)(uintptr_t)gptr)
               : "memory");
}
__device__ __forceinline__ void wait_async0() {
  asm volatile("s_wait_asynccnt 0x0" ::: "memory");
}

// A fragment (16x32, MxK) from row-major u16 LDS [rows][stride], pairs along K.
// Lane l holds row m0+(l&15); VGPR j holds K = (j>=4?16:0) + (l>=16?8:0) + 2*(j&3).
__device__ __forceinline__ bf16x16 load_frag_A(const unsigned short* lds,
                                               int stride, int m0, int k0,
                                               int lane) {
  const unsigned short* row = lds + (m0 + (lane & 15)) * stride + k0;
  const int khalf = (lane >> 4) << 3;
  i32x8 t;
#pragma unroll
  for (int j = 0; j < 8; ++j) {
    int k = ((j >> 2) << 4) + khalf + ((j & 3) << 1);
    t[j] = *(const int*)(row + k);
  }
  return __builtin_bit_cast(bf16x16, t);
}

// B fragment (32x16, KxN) from a TRANSPOSED u16 LDS image [N][stride] so that
// K-pairs are contiguous. Lane l holds col n0+(l&15);
// VGPR j holds K = (j>=4?16:0) + (l>=16?8:0) + 2*(j&3).
__device__ __forceinline__ bf16x16 load_frag_B(const unsigned short* ldsT,
                                               int stride, int n0, int k0,
                                               int lane) {
  const unsigned short* row =
      ldsT + (n0 + (lane & 15)) * stride + k0 + ((lane >> 4) << 3);
  i32x8 t;
#pragma unroll
  for (int j = 0; j < 8; ++j) t[j] = *(const int*)(row + ((j >> 2) << 4) + ((j & 3) << 1));
  return __builtin_bit_cast(bf16x16, t);
}

// ---------------------------------------------------------------------------
// f32 -> bf16 conversion (vectorized by 4)
// ---------------------------------------------------------------------------
__global__ void cvt_f32_bf16(const float* __restrict__ in,
                             unsigned short* __restrict__ out, int n4) {
  int i = blockIdx.x * blockDim.x + threadIdx.x;
  if (i < n4) {
    float4 f = ((const float4*)in)[i];
    uint2 p;
    p.x = (unsigned)f2bf(f.x) | ((unsigned)f2bf(f.y) << 16);
    p.y = (unsigned)f2bf(f.z) | ((unsigned)f2bf(f.w) << 16);
    *(uint2*)(out + 4 * (size_t)i) = p;
  }
}

// ---------------------------------------------------------------------------
// Generic bf16 GEMM: C[M,N] = A[M,K] * B[K,N] (+bias) (+relu), out f32 or bf16
// Tile 128x128, BK=64 (two K=32 WMMA steps per stage), 256 threads = 8 waves
// arranged 2(M) x 4(N); each wave owns a 64x32 output block = 4x2 subtiles.
// A staged via async global->LDS b128 copies; epilogue specialized at compile
// time (no runtime flag branches).
// ---------------------------------------------------------------------------
template <bool RELU, bool OBF16, bool BIAS>
__global__ __launch_bounds__(256) void gemm_bf16(
    const unsigned short* __restrict__ A, const unsigned short* __restrict__ B,
    const float* __restrict__ bias, void* __restrict__ Cout, int M, int N,
    int K) {
  __shared__ unsigned short a_lds[128 * 72];  // [row][k], stride 72 (16B align)
  __shared__ unsigned short b_lds[128 * 72];  // transposed: [n][k], stride 72
  const int tid = threadIdx.x;
  const int lane = tid & 31;
  const int wave = tid >> 5;
  const int bm = blockIdx.y * 128;
  const int bn = blockIdx.x * 128;
  const int wm = (wave >> 2) * 64;
  const int wn = (wave & 3) * 32;

  f32x8 acc[4][2];
#pragma unroll
  for (int mi = 0; mi < 4; ++mi)
#pragma unroll
    for (int ni = 0; ni < 2; ++ni) acc[mi][ni] = zero8();

  for (int k0 = 0; k0 < K; k0 += 64) {
    __syncthreads();  // protect LDS reuse from previous stage's readers
    // stage A 128x64: per-lane async 16B global->LDS copies (ASYNCcnt)
#pragma unroll
    for (int it = 0; it < 4; ++it) {
      int id = it * 256 + tid;    // 0..1023
      int r = id >> 3;            // 0..127
      int kc = (id & 7) << 3;     // 0..56 step 8 (u16 units)
      const unsigned short* gp = A + (size_t)(bm + r) * K + k0 + kc;
      unsigned lds_off = (unsigned)(uintptr_t)(a_lds + r * 72 + kc);
      async_copy_b128(lds_off, gp);
    }
    // stage B 64x128, transposed into [n][k] through VGPRs
#pragma unroll
    for (int it = 0; it < 4; ++it) {
      int id = it * 256 + tid;    // 0..1023
      int kr = id >> 4;           // 0..63
      int nc = (id & 15) << 3;    // 0..120 step 8
      uint4 d = *(const uint4*)(B + (size_t)(k0 + kr) * N + bn + nc);
      unsigned short* bp = b_lds + nc * 72 + kr;
      bp[0 * 72] = (unsigned short)(d.x & 0xFFFFu);
      bp[1 * 72] = (unsigned short)(d.x >> 16);
      bp[2 * 72] = (unsigned short)(d.y & 0xFFFFu);
      bp[3 * 72] = (unsigned short)(d.y >> 16);
      bp[4 * 72] = (unsigned short)(d.z & 0xFFFFu);
      bp[5 * 72] = (unsigned short)(d.z >> 16);
      bp[6 * 72] = (unsigned short)(d.w & 0xFFFFu);
      bp[7 * 72] = (unsigned short)(d.w >> 16);
    }
    wait_async0();   // A-tile async copies complete (this wave's)
    __syncthreads(); // all waves' tiles visible
#pragma unroll
    for (int kk = 0; kk < 2; ++kk) {
      const int ks = kk * 32;
      bf16x16 bf0 = load_frag_B(b_lds, 72, wn + 0, ks, lane);
      bf16x16 bf1 = load_frag_B(b_lds, 72, wn + 16, ks, lane);
#pragma unroll
      for (int mi = 0; mi < 4; ++mi) {
        bf16x16 af = load_frag_A(a_lds, 72, wm + mi * 16, ks, lane);
        acc[mi][0] = wmma_bf16(af, bf0, acc[mi][0]);
        acc[mi][1] = wmma_bf16(af, bf1, acc[mi][1]);
      }
    }
  }

#pragma unroll
  for (int mi = 0; mi < 4; ++mi) {
#pragma unroll
    for (int ni = 0; ni < 2; ++ni) {
      const int col = bn + wn + ni * 16 + (lane & 15);
      const int rbase = bm + wm + mi * 16 + ((lane >> 4) << 3);
      const float badd = BIAS ? bias[col] : 0.0f;
#pragma unroll
      for (int j = 0; j < 8; ++j) {
        float v = acc[mi][ni][j] + badd;
        if (RELU) v = fmaxf(v, 0.0f);
        const size_t idx = (size_t)(rbase + j) * N + col;
        if (OBF16)
          ((unsigned short*)Cout)[idx] = f2bf(v);
        else
          ((float*)Cout)[idx] = v;
      }
    }
  }
}

// ---------------------------------------------------------------------------
// LSH hashing + per-(b,h) bitonic sort of key = (bucket<<12)|pos over S=4096.
// One workgroup (512 threads) per (b,h). Keys are unique -> exactly matches
// the stable argsort(bucket*s + pos) of the reference.
// ---------------------------------------------------------------------------
__global__ __launch_bounds__(512) void hash_sort(
    const float* __restrict__ qk, const float* __restrict__ rot,
    unsigned* __restrict__ skey) {
  __shared__ float rot_s[64 * 32];
  __shared__ unsigned keys[4096];
  const int bh = blockIdx.x;  // b*8 + h
  const int b = bh >> 3, h = bh & 7;
  const int tid = threadIdx.x;

  for (int i = tid; i < 2048; i += 512) rot_s[i] = rot[i];
  __syncthreads();

  for (int p = tid; p < 4096; p += 512) {
    const float* row = qk + ((size_t)(b * 4096 + p) * 512 + h * 64);
    float acc[32];
#pragma unroll
    for (int r = 0; r < 32; ++r) acc[r] = 0.0f;
    for (int d = 0; d < 64; ++d) {
      float q = row[d];
#pragma unroll
      for (int r = 0; r < 32; ++r) acc[r] += q * rot_s[d * 32 + r];
    }
    float best = acc[0];
    int bkt = 0;
#pragma unroll
    for (int r = 1; r < 32; ++r)
      if (acc[r] > best) { best = acc[r]; bkt = r; }
#pragma unroll
    for (int r = 0; r < 32; ++r)
      if (-acc[r] > best) { best = -acc[r]; bkt = 32 + r; }
    keys[p] = ((unsigned)bkt << 12) | (unsigned)p;
  }
  __syncthreads();

  for (int k = 2; k <= 4096; k <<= 1) {
    for (int j = k >> 1; j > 0; j >>= 1) {
      for (int t = tid; t < 2048; t += 512) {
        int low = ((t & ~(j - 1)) << 1) | (t & (j - 1));
        int hi = low + j;
        unsigned a = keys[low], c = keys[hi];
        bool up = ((low & k) == 0);
        if ((a > c) == up) { keys[low] = c; keys[hi] = a; }
      }
      __syncthreads();
    }
  }
  for (int i = tid; i < 4096; i += 512)
    skey[(size_t)bh * 4096 + i] = keys[i];
}

// ---------------------------------------------------------------------------
// Chunked LSH attention. One WG (128 threads / 4 waves) per (b,h,chunk).
// ext rows = [chunk c (64) | chunk c-1 (64)]. Scores = q @ k_ext^T (WMMA),
// bucket-mask + softmax in LDS, O = probs @ v_ext (WMMA), scatter to original
// positions (the unsort).
// ---------------------------------------------------------------------------
__global__ __launch_bounds__(128) void lsh_attn(
    const float* __restrict__ qk, const unsigned short* __restrict__ vbf,
    const unsigned* __restrict__ skey, unsigned short* __restrict__ out) {
  extern __shared__ char smem[];
  unsigned short* q_s = (unsigned short*)smem;        // [64][72] raw q (bf16)
  unsigned short* k_s = q_s + 64 * 72;                // [128][72] normalized k
  unsigned short* vt_s = k_s + 128 * 72;              // [64][136] v transposed
  unsigned short* p_s = vt_s + 64 * 136;              // [64][136] probs (bf16)
  float* sc_s = (float*)(p_s + 64 * 136);             // [64][132] scores (f32)
  int* pos_s = (int*)(sc_s + 64 * 132);               // [128]
  int* bkt_s = pos_s + 128;                           // [128]

  const int tid = threadIdx.x, lane = tid & 31, wave = tid >> 5;
  const int gid = blockIdx.x;  // b*512 + h*64 + c
  const int c = gid & 63;
  const int h = (gid >> 6) & 7;
  const int b = gid >> 9;
  const int cprev = (c + 63) & 63;

  // ---- gather: each thread owns one ext row ----
  {
    const int src_chunk = (tid < 64) ? c : cprev;
    const int i_in = tid & 63;
    unsigned key =
        skey[(size_t)(b * 8 + h) * 4096 + src_chunk * 64 + i_in];
    int pos = (int)(key & 4095u);
    pos_s[tid] = pos;
    bkt_s[tid] = (int)(key >> 12);

    const float* qrow = qk + ((size_t)(b * 4096 + pos) * 512 + h * 64);
    float vals[64];
    float ss = 0.0f;
#pragma unroll
    for (int d = 0; d < 64; d += 4) {
      float4 f = *(const float4*)(qrow + d);
      vals[d] = f.x; vals[d + 1] = f.y; vals[d + 2] = f.z; vals[d + 3] = f.w;
      ss += f.x * f.x + f.y * f.y + f.z * f.z + f.w * f.w;
    }
    float inv = 1.0f / (sqrtf(ss) + 1e-9f);
#pragma unroll
    for (int d = 0; d < 64; ++d) {
      k_s[tid * 72 + d] = f2bf(vals[d] * inv);
      if (tid < 64) q_s[tid * 72 + d] = f2bf(vals[d]);
    }
    const unsigned short* vrow =
        vbf + ((size_t)(b * 4096 + pos) * 512 + h * 64);
#pragma unroll
    for (int d = 0; d < 64; d += 2) {
      unsigned pr = *(const unsigned*)(vrow + d);
      vt_s[(d + 0) * 136 + tid] = (unsigned short)(pr & 0xFFFFu);
      vt_s[(d + 1) * 136 + tid] = (unsigned short)(pr >> 16);
    }
  }
  __syncthreads();

  // ---- scores 64x128 = q[64x64] @ k_ext^T ; wave owns 32 score columns ----
#pragma unroll
  for (int nt = 0; nt < 2; ++nt) {
    int n0 = wave * 32 + nt * 16;
#pragma unroll
    for (int mt = 0; mt < 4; ++mt) {
      f32x8 acc = zero8();
#pragma unroll
      for (int ks = 0; ks < 64; ks += 32) {
        bf16x16 af = load_frag_A(q_s, 72, mt * 16, ks, lane);
        bf16x16 bf = load_frag_B(k_s, 72, n0, ks, lane);
        acc = wmma_bf16(af, bf, acc);
      }
      int col = n0 + (lane & 15);
      int rbase = mt * 16 + ((lane >> 4) << 3);
#pragma unroll
      for (int j = 0; j < 8; ++j) sc_s[(rbase + j) * 132 + col] = acc[j];
    }
  }
  __syncthreads();

  // ---- mask + softmax: one thread per query row ----
  if (tid < 64) {
    const int mybkt = bkt_s[tid];
    float* srow = sc_s + tid * 132;
    float mx = -1e30f;
    for (int kk = 0; kk < 128; ++kk) {
      float v = (bkt_s[kk] == mybkt) ? srow[kk] * 0.125f : -1e9f;
      srow[kk] = v;
      mx = fmaxf(mx, v);
    }
    float sum = 0.0f;
    for (int kk = 0; kk < 128; ++kk) {
      float e = __expf(srow[kk] - mx);
      srow[kk] = e;
      sum += e;
    }
    float inv = 1.0f / sum;
    unsigned short* prow = p_s + tid * 136;
    for (int kk = 0; kk < 128; ++kk) prow[kk] = f2bf(srow[kk] * inv);
  }
  __syncthreads();

  // ---- O[64x64] = probs[64x128] @ v_ext[128x64] ; wave owns 16 out cols ----
  {
    int n0 = wave * 16;
#pragma unroll
    for (int mt = 0; mt < 4; ++mt) {
      f32x8 acc = zero8();
#pragma unroll
      for (int ks = 0; ks < 128; ks += 32) {
        bf16x16 af = load_frag_A(p_s, 136, mt * 16, ks, lane);
        bf16x16 bf = load_frag_B(vt_s, 136, n0, ks, lane);
        acc = wmma_bf16(af, bf, acc);
      }
      int col = n0 + (lane & 15);
      int rbase = mt * 16 + ((lane >> 4) << 3);
#pragma unroll
      for (int j = 0; j < 8; ++j) {
        int pos = pos_s[rbase + j];  // unsort: scatter to original position
        out[(size_t)(b * 4096 + pos) * 512 + h * 64 + col] = f2bf(acc[j]);
      }
    }
  }
}

// ---------------------------------------------------------------------------
// y = LayerNorm(a + res) * g + beta ; optionally emits a bf16 copy.
// One 256-thread block per row of 512; in-place safe (outf may alias a).
// ---------------------------------------------------------------------------
__global__ __launch_bounds__(256) void resid_ln(
    const float* __restrict__ a, const float* __restrict__ res,
    const float* __restrict__ g, const float* __restrict__ beta,
    float* __restrict__ outf, unsigned short* __restrict__ outbf) {
  __shared__ float r1[256];
  __shared__ float r2[256];
  const int row = blockIdx.x, tid = threadIdx.x;
  const float* ar = a + (size_t)row * 512;
  const float* rr = res + (size_t)row * 512;
  float v0 = ar[tid] + rr[tid];
  float v1 = ar[tid + 256] + rr[tid + 256];
  r1[tid] = v0 + v1;
  r2[tid] = v0 * v0 + v1 * v1;
  __syncthreads();
  for (int s = 128; s > 0; s >>= 1) {
    if (tid < s) { r1[tid] += r1[tid + s]; r2[tid] += r2[tid + s]; }
    __syncthreads();
  }
  float mean = r1[0] * (1.0f / 512.0f);
  float var = r2[0] * (1.0f / 512.0f) - mean * mean;
  float inv = rsqrtf(var + 1e-5f);
  float y0 = (v0 - mean) * inv * g[tid] + beta[tid];
  float y1 = (v1 - mean) * inv * g[tid + 256] + beta[tid + 256];
  float* of = outf + (size_t)row * 512;
  of[tid] = y0;
  of[tid + 256] = y1;
  if (outbf) {
    unsigned short* ob = outbf + (size_t)row * 512;
    ob[tid] = f2bf(y0);
    ob[tid + 256] = f2bf(y1);
  }
}

// ---------------------------------------------------------------------------
// Host orchestration
// ---------------------------------------------------------------------------
extern "C" void kernel_launch(void* const* d_in, const int* in_sizes, int n_in,
                              void* d_out, int out_size, void* d_ws,
                              size_t ws_size, hipStream_t stream) {
  (void)in_sizes; (void)n_in; (void)out_size; (void)ws_size;
  const float* x = (const float*)d_in[0];
  const float* Wqk = (const float*)d_in[1];
  const float* Wv = (const float*)d_in[2];
  const float* Wo = (const float*)d_in[3];
  const float* rot = (const float*)d_in[4];
  const float* ln1g = (const float*)d_in[5];
  const float* ln1b = (const float*)d_in[6];
  const float* ln2g = (const float*)d_in[7];
  const float* ln2b = (const float*)d_in[8];
  const float* W1 = (const float*)d_in[9];
  const float* b1 = (const float*)d_in[10];
  const float* W2 = (const float*)d_in[11];
  const float* b2 = (const float*)d_in[12];
  float* out = (float*)d_out;

  const size_t NT = 32768;  // B*S tokens
  char* ws = (char*)d_ws;
  size_t off = 0;
  auto carve = [&](size_t bytes) {
    size_t o = off;
    off += (bytes + 255) & ~(size_t)255;
    return o;
  };
  unsigned short* xbf   = (unsigned short*)(ws + carve(NT * 512 * 2));
  unsigned short* wqkbf = (unsigned short*)(ws + carve(512 * 512 * 2));
  unsigned short* wvbf  = (unsigned short*)(ws + carve(512 * 512 * 2));
  unsigned short* wobf  = (unsigned short*)(ws + carve(512 * 512 * 2));
  unsigned short* w1bf  = (unsigned short*)(ws + carve(512 * 2048 * 2));
  unsigned short* w2bf  = (unsigned short*)(ws + carve(2048 * 512 * 2));
  float* qkbuf          = (float*)(ws + carve(NT * 512 * 4));  // qk -> o -> x1
  unsigned short* vbf   = (unsigned short*)(ws + carve(NT * 512 * 2));
  unsigned* skey        = (unsigned*)(ws + carve(64 * 4096 * 4));
  unsigned short* attnb = (unsigned short*)(ws + carve(NT * 512 * 2));
  unsigned short* hbf   = (unsigned short*)(ws + carve(NT * 2048 * 2));

  // 1) bf16 conversions
  cvt_f32_bf16<<<(NT * 512 / 4 + 255) / 256, 256, 0, stream>>>(x, xbf,
                                                               NT * 512 / 4);
  cvt_f32_bf16<<<(512 * 512 / 4 + 255) / 256, 256, 0, stream>>>(Wqk, wqkbf,
                                                                512 * 512 / 4);
  cvt_f32_bf16<<<(512 * 512 / 4 + 255) / 256, 256, 0, stream>>>(Wv, wvbf,
                                                                512 * 512 / 4);
  cvt_f32_bf16<<<(512 * 512 / 4 + 255) / 256, 256, 0, stream>>>(Wo, wobf,
                                                                512 * 512 / 4);
  cvt_f32_bf16<<<(512 * 2048 / 4 + 255) / 256, 256, 0, stream>>>(
      W1, w1bf, 512 * 2048 / 4);
  cvt_f32_bf16<<<(2048 * 512 / 4 + 255) / 256, 256, 0, stream>>>(
      W2, w2bf, 2048 * 512 / 4);

  // 2) qk = x @ Wqk (f32 out for hashing/normalization); v = x @ Wv (bf16 out)
  gemm_bf16<false, false, false><<<dim3(512 / 128, NT / 128), 256, 0, stream>>>(
      xbf, wqkbf, nullptr, qkbuf, (int)NT, 512, 512);
  gemm_bf16<false, true, false><<<dim3(512 / 128, NT / 128), 256, 0, stream>>>(
      xbf, wvbf, nullptr, vbf, (int)NT, 512, 512);

  // 3) LSH hash + stable sort per (b,h)
  hash_sort<<<64, 512, 0, stream>>>(qkbuf, rot, skey);

  // 4) chunked attention (B*H*C = 4096 workgroups), ~95KB dynamic LDS
  const size_t attn_lds =
      (64 * 72 + 128 * 72 + 64 * 136 + 64 * 136) * 2 + 64 * 132 * 4 +
      128 * 4 * 2;
  lsh_attn<<<4096, 128, attn_lds, stream>>>(qkbuf, vbf, skey, attnb);

  // 5) o = attn @ Wo (reuse qk buffer)
  gemm_bf16<false, false, false><<<dim3(512 / 128, NT / 128), 256, 0, stream>>>(
      attnb, wobf, nullptr, qkbuf, (int)NT, 512, 512);

  // 6) x1 = LN(o + x) in-place, bf16 copy into xbf
  resid_ln<<<NT, 256, 0, stream>>>(qkbuf, x, ln1g, ln1b, qkbuf, xbf);

  // 7) h = relu(x1 @ W1 + b1) (bf16 out)
  gemm_bf16<true, true, true><<<dim3(2048 / 128, NT / 128), 256, 0, stream>>>(
      xbf, w1bf, b1, hbf, (int)NT, 2048, 512);

  // 8) ff = h @ W2 + b2 -> d_out (f32)
  gemm_bf16<false, false, true><<<dim3(512 / 128, NT / 128), 256, 0, stream>>>(
      hbf, w2bf, b2, out, (int)NT, 512, 2048);

  // 9) out = LN(ff + x1) in-place
  resid_ln<<<NT, 256, 0, stream>>>(out, qkbuf, ln2g, ln2b, out, nullptr);
}
